// OuterProductMean_72627896975602
// MI455X (gfx1250) — matrix operand order repaired and checked by reading, not compiled, not run on hardware
//
#include <hip/hip_runtime.h>
#include <hip/hip_bf16.h>

// ---------------------------------------------------------------------------
// OuterProductMean for MI455X (gfx1250, wave32, WMMA 16x16x32 f16)
//   N=4, L=384, C_S=384, C_H=32, C_Z=128
// Factorized:  T[r,z,d] = sum_c a[r,c] * Wo3[c,d,z]   (r = n*L+i)
//              out[n,i,j,z] = sum_d b[n*L+j,d] * T[n*L+i,z,d] + bo[z]
// Both contractions have K=32 -> exactly one v_wmma_f32_16x16x32_f16 per tile.
// ---------------------------------------------------------------------------

#define OPM_N   4
#define OPM_L   384
#define OPM_CS  384
#define OPM_CH  32
#define OPM_CZ  128
#define OPM_NL  (OPM_N * OPM_L)         // 1536
#define OPM_CC  (OPM_CH * OPM_CZ)       // 4096 (flattened (z,d) with cc = z*32+d)

typedef _Float16 v16h  __attribute__((ext_vector_type(16)));
typedef _Float16 half8 __attribute__((ext_vector_type(8)));
typedef float    v8f   __attribute__((ext_vector_type(8)));

#if __has_builtin(__builtin_amdgcn_global_load_async_to_lds_b128) && \
    __has_builtin(__builtin_amdgcn_s_wait_asynccnt)
#define OPM_ASYNC_LDS 1
#endif

// -------------------------------------------------------------------
// Kernel 1: fused LayerNorm + two projections. One block per row.
//   a16[row*32+h] = (sn @ W1 + b1)[h],  b16 likewise with W2/b2 (f16)
// -------------------------------------------------------------------
__global__ __launch_bounds__(128) void k_ln_proj(
    const float* __restrict__ s,
    const float* __restrict__ ln_g, const float* __restrict__ ln_b,
    const float* __restrict__ W1, const float* __restrict__ b1,
    const float* __restrict__ W2, const float* __restrict__ b2,
    _Float16* __restrict__ a16, _Float16* __restrict__ b16)
{
  const int row = blockIdx.x;       // 0..1535
  const int t   = threadIdx.x;      // 0..127
  __shared__ float sn[OPM_CS];
  __shared__ float red[128];

  const float* srow = s + (size_t)row * OPM_CS;
  float v0 = srow[t], v1 = srow[t + 128], v2 = srow[t + 256];

  red[t] = v0 + v1 + v2;
  __syncthreads();
  for (int off = 64; off > 0; off >>= 1) {
    if (t < off) red[t] += red[t + off];
    __syncthreads();
  }
  const float mu = red[0] * (1.0f / OPM_CS);
  __syncthreads();

  red[t] = v0 * v0 + v1 * v1 + v2 * v2;
  __syncthreads();
  for (int off = 64; off > 0; off >>= 1) {
    if (t < off) red[t] += red[t + off];
    __syncthreads();
  }
  const float var = red[0] * (1.0f / OPM_CS) - mu * mu;
  const float rs  = __frsqrt_rn(var + 1e-5f);
  __syncthreads();

  sn[t]       = (v0 - mu) * rs * ln_g[t]       + ln_b[t];
  sn[t + 128] = (v1 - mu) * rs * ln_g[t + 128] + ln_b[t + 128];
  sn[t + 256] = (v2 - mu) * rs * ln_g[t + 256] + ln_b[t + 256];
  __syncthreads();

  if (t < 64) {
    const int   h  = t & 31;
    const float* W  = (t < 32) ? W1 : W2;
    const float* bb = (t < 32) ? b1 : b2;
    float acc = bb[h];
    #pragma unroll 4
    for (int c = 0; c < OPM_CS; ++c) acc += sn[c] * W[c * OPM_CH + h];
    _Float16* dst = (t < 32) ? a16 : b16;
    dst[row * OPM_CH + h] = (_Float16)acc;
  }
}

// -------------------------------------------------------------------
// Kernel 2: Wo (f32, [(c*32+d), z]) -> wo16 (f16, [cc=z*32+d][c])
// -------------------------------------------------------------------
__global__ __launch_bounds__(256) void k_wo_cvt(
    const float* __restrict__ Wo, _Float16* __restrict__ wo16)
{
  const int idx = blockIdx.x * 256 + threadIdx.x;  // 0..131071
  const int c  = idx & 31;
  const int cc = idx >> 5;
  const int d  = cc & 31;
  const int z  = cc >> 5;
  wo16[idx] = (_Float16)Wo[(size_t)(c * OPM_CH + d) * OPM_CZ + z];
}

// -------------------------------------------------------------------
// Kernel 3: T16[row][cc] = sum_c a16[row,c] * wo16[cc,c]
//   GEMM [1536,32] x [32,4096], one wmma per 16x16 tile (K=32).
//   grid = (96 row tiles, 16 col slabs of 256), 8 waves * 2 tiles.
// -------------------------------------------------------------------
__global__ __launch_bounds__(256) void k_build_T(
    const _Float16* __restrict__ a16,
    const _Float16* __restrict__ wo16,
    _Float16* __restrict__ T16)
{
  const int lane = threadIdx.x & 31;
  const int wv   = threadIdx.x >> 5;      // 0..7
  const int ml   = lane & 15;
  const int hi   = lane >> 4;
  const int r0   = blockIdx.x * 16;       // row tile base
  const int cb   = blockIdx.y * 256;      // col slab base

  // A fragment: rows r0+ml, k = hi*16 + p
  const _Float16* arow = a16 + (size_t)(r0 + ml) * OPM_CH;
  half8 alo = *(const half8*)(arow + hi * 16);
  half8 ahi = *(const half8*)(arow + hi * 16 + 8);
  v16h A;
  #pragma unroll
  for (int p = 0; p < 8; ++p) { A[p] = alo[p]; A[p + 8] = ahi[p]; }

  #pragma unroll
  for (int tt = 0; tt < 2; ++tt) {
    const int cc0 = cb + (wv * 2 + tt) * 16;
    // B fragment: col cc0+ml, k = hi*16 + p (contiguous in wo16 [cc][c])
    const _Float16* bcol = wo16 + (size_t)(cc0 + ml) * OPM_CH + hi * 16;
    half8 blo = *(const half8*)(bcol);
    half8 bhi = *(const half8*)(bcol + 8);
    v16h B;
    #pragma unroll
    for (int p = 0; p < 8; ++p) { B[p] = blo[p]; B[p + 8] = bhi[p]; }

    v8f acc = {};
    acc = __builtin_amdgcn_wmma_f32_16x16x32_f16(
        false, A, false, B, (short)0, acc, false, false);

    #pragma unroll
    for (int r = 0; r < 8; ++r)
      T16[(size_t)(r0 + r + hi * 8) * OPM_CC + cc0 + ml] = (_Float16)acc[r];
  }
}

// -------------------------------------------------------------------
// Kernel 4: out[n,i,j,z] = sum_d b16[n*L+j,d] * T16[n*L+i][z*32+d] + bo[z]
//   grid = (3 j-slabs of 128, i=384, n=4); 8 waves, 16 j each, 8 z-tiles.
//   T slice (8KB) staged in LDS via CDNA5 async-to-LDS path.
// -------------------------------------------------------------------
__global__ __launch_bounds__(256) void k_outer(
    const _Float16* __restrict__ b16,
    const _Float16* __restrict__ T16,
    const float* __restrict__ bo,
    float* __restrict__ out)
{
  const int n    = blockIdx.z;
  const int i    = blockIdx.y;
  const int js   = blockIdx.x;            // 0..2
  const int tid  = threadIdx.x;
  const int lane = tid & 31;
  const int wv   = tid >> 5;              // 0..7
  const int ml   = lane & 15;
  const int hi   = lane >> 4;

  __shared__ alignas(16) _Float16 Ts[OPM_CC];   // [z][d], 8 KB

  const _Float16* Trow = T16 + (size_t)(n * OPM_L + i) * OPM_CC;
#ifdef OPM_ASYNC_LDS
  {
    // Builtin signature (from clang diagnostic): first param is
    // "int __attribute__((vector_size(16))) __device__ *"  i.e. int4 AS(1)*.
    typedef int opm_i4 __attribute__((vector_size(16)));
    typedef __attribute__((address_space(1))) opm_i4 opm_g4;
    typedef __attribute__((address_space(3))) opm_i4 opm_l4;
    const char* g = (const char*)Trow;
    const char* l = (const char*)&Ts[0];
    // Flat LDS pointers carry the LDS offset in their low 32 bits, so the
    // integer truncation recovers the addrspace(3) pointer value.
    __builtin_amdgcn_global_load_async_to_lds_b128(
        (opm_g4*)(g + tid * 16),
        (opm_l4*)(unsigned)(size_t)(l + tid * 16), 0, 0);
    __builtin_amdgcn_global_load_async_to_lds_b128(
        (opm_g4*)(g + 4096 + tid * 16),
        (opm_l4*)(unsigned)(size_t)(l + 4096 + tid * 16), 0, 0);
    __builtin_amdgcn_s_wait_asynccnt(0);
  }
#else
  {
    const half8* src = (const half8*)Trow;
    half8*       dst = (half8*)Ts;
    dst[tid]       = src[tid];
    dst[tid + 256] = src[tid + 256];
  }
#endif
  __syncthreads();

  // A fragment: 16 rows of b (j = j0+ml), K = 32 d-values
  const int j0 = js * 128 + wv * 16;
  const _Float16* brow = b16 + (size_t)(n * OPM_L + j0 + ml) * OPM_CH;
  half8 alo = *(const half8*)(brow + hi * 16);
  half8 ahi = *(const half8*)(brow + hi * 16 + 8);
  v16h A;
  #pragma unroll
  for (int p = 0; p < 8; ++p) { A[p] = alo[p]; A[p + 8] = ahi[p]; }

  float* obase = out + ((size_t)(n * OPM_L + i) * OPM_L + j0) * OPM_CZ;

  #pragma unroll
  for (int zt = 0; zt < 8; ++zt) {
    const int z0 = zt * 16;
    // B fragment: col z = z0+ml, k = hi*16+p -> contiguous in Ts[z][d]
    const _Float16* tcol = Ts + (z0 + ml) * OPM_CH + hi * 16;
    half8 blo = *(const half8*)(tcol);
    half8 bhi = *(const half8*)(tcol + 8);
    v16h B;
    #pragma unroll
    for (int p = 0; p < 8; ++p) { B[p] = blo[p]; B[p + 8] = bhi[p]; }

    const float bz = bo[z0 + ml];
    v8f acc;
    #pragma unroll
    for (int r = 0; r < 8; ++r) acc[r] = bz;

    acc = __builtin_amdgcn_wmma_f32_16x16x32_f16(
        false, A, false, B, (short)0, acc, false, false);

    float* ocol = obase + z0 + ml;
    #pragma unroll
    for (int r = 0; r < 8; ++r)
      ocol[(size_t)(r + hi * 8) * OPM_CZ] = acc[r];
  }
}

// -------------------------------------------------------------------
extern "C" void kernel_launch(void* const* d_in, const int* in_sizes, int n_in,
                              void* d_out, int out_size, void* d_ws, size_t ws_size,
                              hipStream_t stream) {
  (void)in_sizes; (void)n_in; (void)out_size; (void)ws_size;
  const float* s    = (const float*)d_in[0];
  const float* ln_g = (const float*)d_in[1];
  const float* ln_b = (const float*)d_in[2];
  const float* W1   = (const float*)d_in[3];
  const float* b1   = (const float*)d_in[4];
  const float* W2   = (const float*)d_in[5];
  const float* b2   = (const float*)d_in[6];
  const float* Wo   = (const float*)d_in[7];
  const float* bo   = (const float*)d_in[8];
  float* out = (float*)d_out;

  // Workspace layout (bytes): wo16 262144 | a16 98304 | b16 98304 | T16 12582912
  char* ws = (char*)d_ws;
  _Float16* wo16 = (_Float16*)(ws);
  _Float16* a16  = (_Float16*)(ws + 262144);
  _Float16* b16  = (_Float16*)(ws + 262144 + 98304);
  _Float16* T16  = (_Float16*)(ws + 262144 + 196608);

  k_ln_proj<<<OPM_NL, 128, 0, stream>>>(s, ln_g, ln_b, W1, b1, W2, b2, a16, b16);
  k_wo_cvt<<<(OPM_CC * OPM_CH) / 256, 256, 0, stream>>>(Wo, wo16);
  k_build_T<<<dim3(OPM_NL / 16, OPM_CC / 256), 256, 0, stream>>>(a16, wo16, T16);
  k_outer<<<dim3(OPM_L / 128, OPM_L, OPM_N), 256, 0, stream>>>(b16, T16, bo, out);
}
